// TimeSeriesCircularConvolutionPositionalEncoding_69346541961448
// MI455X (gfx1250) — compile-verified, matrix operand order checked by reading, not run on hardware
//
#include <hip/hip_runtime.h>
#include <stdint.h>

// CDNA5 WMMA types
typedef __attribute__((ext_vector_type(16))) __bf16        v16bf;
typedef __attribute__((ext_vector_type(8)))  float         v8f;
typedef __attribute__((ext_vector_type(4)))  unsigned int  u32x4;
typedef __attribute__((ext_vector_type(4)))  float         f32x4;

#define SEQ  4096
#define DIM  512
#define BSZ  16
#define RROW 1040   // shifted reversed-pos row length (1040*2B = 2080B, multiple of 16)

__device__ __forceinline__ unsigned short f32_to_bf16(float f) {
  union { float f; uint32_t u; } c; c.f = f;
  uint32_t u = c.u;
  uint32_t r = u + 0x7FFFu + ((u >> 16) & 1u);   // round-to-nearest-even
  return (unsigned short)(r >> 16);
}

union Frag16 {
  v16bf v;
  u32x4 q[2];   // q[0] = elements 0..7 (K=kh..kh+7), q[1] = elements 8..15 (K=kh+16..kh+23)
};

__device__ __forceinline__ v8f wmma_bf16(const Frag16& a, const Frag16& b, v8f c) {
  return __builtin_amdgcn_wmma_f32_16x16x32_bf16(
      false, a.v, false, b.v, (short)0, c, false, false);
}

// one workgroup per sequence position s; 8 waves; wave w owns output cols [64w, 64w+64)
__global__ __launch_bounds__(256)
void circconv_wmma_kernel(const float* __restrict__ x,     // [16][4096][512]
                          const float* __restrict__ pos,   // [4096][512]
                          float* __restrict__ out) {       // [16][4096][512]
  __shared__ __align__(16) unsigned short Xl[BSZ][DIM];    // X_s tile, bf16 (16 KB)
  // 8 shift copies of the reversed position row: Rbig[c][t] = p[(1024 - t - c) & 511]
  __shared__ __align__(16) unsigned short Rbig[8][RROW];   // 16.25 KB

  const int s   = blockIdx.x;
  const int tid = threadIdx.x;

  // ---- stage X_s (16x512 f32 -> bf16 in LDS): float4 NT loads, 8B packed DS stores ----
  for (int i = tid; i < (BSZ * DIM) / 4; i += 256) {
    const int e4 = i * 4;
    const int b  = e4 >> 9;
    const int d  = e4 & 511;                 // multiple of 4
    const f32x4 v = __builtin_nontemporal_load(
        (const f32x4*)(x + ((size_t)b * SEQ + s) * DIM + d));
    union { unsigned short h[4]; uint64_t u; } pk;
    pk.h[0] = f32_to_bf16(v.x); pk.h[1] = f32_to_bf16(v.y);
    pk.h[2] = f32_to_bf16(v.z); pk.h[3] = f32_to_bf16(v.w);
    *(uint64_t*)&Xl[b][d] = pk.u;            // 8B aligned (d % 4 == 0)
  }
  // ---- stage reversed pos row into 8 shifted copies ----
  for (int t = tid; t < 1032; t += 256) {
    const unsigned short hv = f32_to_bf16(pos[(size_t)s * DIM + ((1024 - t) & 511)]);
    #pragma unroll
    for (int c = 0; c < 8; ++c) {
      if (t >= c) Rbig[c][t - c] = hv;       // Rbig[c][t'] = p[(1024 - t' - c) & 511]
    }
  }
  __syncthreads();

  const int lane = tid & 31;
  const int wv   = tid >> 5;        // wave id 0..7
  const int half = lane >> 4;       // K-half select (A & B 16-bit layouts)
  const int l15  = lane & 15;       // A: row m ; B/D: col within tile
  const int m    = l15;             // batch row for A fragment
  const int kh   = half * 8;

  // B chunk stream (diagonal circulant reuse): all 4 tiles, all K-steps read from
  // one stride-16 stream of b128 chunks anchored at tile 3's offset.
  // C_i = b128 @ pb + 16*i ; tile t at step s uses chunks (3-t)+2s and (4-t)+2s.
  const int n3  = wv * 64 + 48 + l15;        // tile-3 column of this lane
  const int off = 512 - n3;                  // logical start in reversed row
  const int c   = off & 7;                   // shift copy -> 16B-aligned physical offset
  const unsigned short* pb = &Rbig[c][(off - c) + kh];

  v8f acc[4];
  #pragma unroll
  for (int t = 0; t < 4; ++t)
    acc[t] = (v8f){0.f, 0.f, 0.f, 0.f, 0.f, 0.f, 0.f, 0.f};

  // prologue: fill 5-chunk window (i = 0..4)
  u32x4 w0 = *(const u32x4*)(pb +  0);
  u32x4 w1 = *(const u32x4*)(pb + 16);
  u32x4 w2 = *(const u32x4*)(pb + 32);
  u32x4 w3 = *(const u32x4*)(pb + 48);
  u32x4 w4 = *(const u32x4*)(pb + 64);

  #pragma unroll
  for (int st = 0; st < 16; ++st) {
    const int k0 = 32 * st;
    Frag16 a;                                   // A: 2 aligned ds_load_b128
    a.q[0] = *(const u32x4*)&Xl[m][k0 + kh];
    a.q[1] = *(const u32x4*)&Xl[m][k0 + kh + 16];

    Frag16 b;
    b.q[0] = w0; b.q[1] = w1; acc[3] = wmma_bf16(a, b, acc[3]);
    b.q[0] = w1; b.q[1] = w2; acc[2] = wmma_bf16(a, b, acc[2]);
    b.q[0] = w2; b.q[1] = w3; acc[1] = wmma_bf16(a, b, acc[1]);
    b.q[0] = w3; b.q[1] = w4; acc[0] = wmma_bf16(a, b, acc[0]);

    if (st < 15) {                              // slide window by 2 chunks
      w0 = w2; w1 = w3; w2 = w4;
      w3 = *(const u32x4*)(pb + 16 * (2 * st + 5));
      w4 = *(const u32x4*)(pb + 16 * (2 * st + 6));
    }
  }

  // ---- store D tiles: VGPR r -> row (r + 8*half), col = n (documented f32 C/D layout) ----
  #pragma unroll
  for (int t = 0; t < 4; ++t) {
    const int n = wv * 64 + t * 16 + l15;
    #pragma unroll
    for (int r = 0; r < 8; ++r) {
      const int brow = r + half * 8;                       // batch index
      float* p = out + ((size_t)brow * SEQ + s) * DIM + n;
      __builtin_nontemporal_store(acc[t][r], p);           // streamed, never re-read
    }
  }
}

extern "C" void kernel_launch(void* const* d_in, const int* in_sizes, int n_in,
                              void* d_out, int out_size, void* d_ws, size_t ws_size,
                              hipStream_t stream) {
  const float* x   = (const float*)d_in[0];   // (16, 4096, 512) f32
  const float* pos = (const float*)d_in[1];   // (4096, 512) f32
  float* out       = (float*)d_out;           // (16, 4096, 512) f32
  (void)in_sizes; (void)n_in; (void)out_size; (void)d_ws; (void)ws_size;

  dim3 grid(SEQ);     // one workgroup per sequence position
  dim3 block(256);    // 8 waves
  circconv_wmma_kernel<<<grid, block, 0, stream>>>(x, pos, out);
}
